// Qwen3MoeDecoderLayer_21449066676831
// MI455X (gfx1250) — matrix operand order repaired and checked by reading, not compiled, not run on hardware
//
#include <hip/hip_runtime.h>

// ---------------------------------------------------------------------------
// Qwen3-MoE decoder layer for MI455X (gfx1250, wave32, WMMA).
// GEMMs: v_wmma_f32_16x16x32_bf16 fed by double-buffered async LDS staging
// (global_load_async_to_lds_b128 + s_wait_asynccnt).  Attention: flash-style
// WMMA with online softmax.  B=2, S=1024, H=1024, Hq=16, Hkv=4, D=128, I=1024.
// ---------------------------------------------------------------------------

typedef __bf16 bf16;
typedef __attribute__((ext_vector_type(16))) __bf16 v16bf;
typedef __attribute__((ext_vector_type(8)))  __bf16 v8bf;
typedef __attribute__((ext_vector_type(8)))  float  v8f;

#define WMMA_BF16(a, b, c) \
  __builtin_amdgcn_wmma_f32_16x16x32_bf16(false, (a), false, (b), (short)0, (c), false, false)

__device__ __forceinline__ v16bf cat8(v8bf lo, v8bf hi) {
  return __builtin_shufflevector(lo, hi, 0,1,2,3,4,5,6,7,8,9,10,11,12,13,14,15);
}
// A-fragment (16-bit A 16x32): per lane, elements 0..7 -> K = k0+ho+j,
// elements 8..15 -> K = k0+16+ho+(j-8), ho = 8 for lanes 16..31.
__device__ __forceinline__ v16bf load_afrag(const bf16* p) {
  v8bf lo = *(const v8bf*)p;
  v8bf hi = *(const v8bf*)(p + 16);
  return cat8(lo, hi);
}

// Async copy of 16 bytes per lane, global -> LDS (ASYNCcnt-tracked DMA path).
// lds_byte_off is the wave-relative LDS byte address (low 32 bits of a
// generic pointer to __shared__, i.e. the AS3 address the VDST operand wants).
__device__ __forceinline__ void async_copy_b128(unsigned lds_byte_off, const void* gaddr) {
  asm volatile("global_load_async_to_lds_b128 %0, %1, off"
               :: "v"(lds_byte_off), "v"(gaddr) : "memory");
}
__device__ __forceinline__ unsigned lds_off(const void* p) {
  return (unsigned)(unsigned long long)p;   // AS3 addr lives in the low 32 bits
}

__device__ __forceinline__ float wave_sum32(float v) {
#pragma unroll
  for (int off = 16; off > 0; off >>= 1) v += __shfl_xor(v, off, 32);
  return v;
}
__device__ __forceinline__ float half_max16(float v) {
#pragma unroll
  for (int off = 8; off > 0; off >>= 1) v = fmaxf(v, __shfl_xor(v, off, 16));
  return v;
}
__device__ __forceinline__ float half_sum16(float v) {
#pragma unroll
  for (int off = 8; off > 0; off >>= 1) v += __shfl_xor(v, off, 16);
  return v;
}

// ---------------------------------------------------------------------------
// Transpose-convert f32 weight [K,N] (row-major) -> bf16 Wt [N,K] (row-major).
// blockIdx.z batches experts.
// ---------------------------------------------------------------------------
__global__ void transpose_to_bf16(const float* __restrict__ W, bf16* __restrict__ Wt,
                                  int K, int N) {
  size_t total = (size_t)K * N;
  size_t base = (size_t)blockIdx.z * total;
  size_t idx = (size_t)blockIdx.x * blockDim.x + threadIdx.x;
  if (idx >= total) return;
  int n = (int)(idx / K);
  int k = (int)(idx - (size_t)n * K);
  Wt[base + idx] = (bf16)W[base + (size_t)k * N + n];
}

// ---------------------------------------------------------------------------
// RMSNorm row kernel: out_bf16 = x * rsqrt(mean(x^2)+eps) * w.  H = 1024.
// ---------------------------------------------------------------------------
__global__ __launch_bounds__(256)
void rmsnorm_bf16_kernel(const float* __restrict__ x, const float* __restrict__ w,
                         bf16* __restrict__ out, int H) {
  const int t = blockIdx.x;
  const size_t base = (size_t)t * H;
  float rv[4];
  float ssq = 0.f;
#pragma unroll
  for (int j = 0; j < 4; ++j) {
    int i = threadIdx.x + j * 256;
    float v = x[base + i];
    rv[j] = v;
    ssq += v * v;
  }
  ssq = wave_sum32(ssq);
  __shared__ float red[8];
  if ((threadIdx.x & 31) == 0) red[threadIdx.x >> 5] = ssq;
  __syncthreads();
  float tot = 0.f;
#pragma unroll
  for (int k = 0; k < 8; ++k) tot += red[k];
  float inv = rsqrtf(tot / (float)H + 1e-6f);
#pragma unroll
  for (int j = 0; j < 4; ++j) {
    int i = threadIdx.x + j * 256;
    out[base + i] = (bf16)(rv[j] * inv * w[i]);
  }
}

// ---------------------------------------------------------------------------
// resid = hidden + attn_out ; xn = rmsnorm(resid, w)  (H = 1024)
// ---------------------------------------------------------------------------
__global__ __launch_bounds__(256)
void add_resid_rmsnorm_kernel(const float* __restrict__ hid, const float* __restrict__ att,
                              const float* __restrict__ w, float* __restrict__ resid,
                              bf16* __restrict__ xn, int H) {
  const int t = blockIdx.x;
  const size_t base = (size_t)t * H;
  float rv[4];
  float ssq = 0.f;
#pragma unroll
  for (int j = 0; j < 4; ++j) {
    int i = threadIdx.x + j * 256;
    float v = hid[base + i] + att[base + i];
    rv[j] = v;
    resid[base + i] = v;
    ssq += v * v;
  }
  ssq = wave_sum32(ssq);
  __shared__ float red[8];
  if ((threadIdx.x & 31) == 0) red[threadIdx.x >> 5] = ssq;
  __syncthreads();
  float tot = 0.f;
#pragma unroll
  for (int k = 0; k < 8; ++k) tot += red[k];
  float inv = rsqrtf(tot / (float)H + 1e-6f);
#pragma unroll
  for (int j = 0; j < 4; ++j) {
    int i = threadIdx.x + j * 256;
    xn[base + i] = (bf16)(rv[j] * inv * w[i]);
  }
}

// ---------------------------------------------------------------------------
// WMMA GEMM with async LDS double buffering.
//   C[M,N] (f32) = A[M,K] (bf16, row-major) x Bt[N,K]^T.
// Block = 128 threads = 4 waves.  Block tile 64x64; wave w owns rows
// [64*by + 16*w, +16) x all 64 cols (4 accumulators).  K-step = 32.
// Stage: A tile 64x32 (4KB) + B tile 64x32 (4KB) via 4 async b128 per wave,
// double buffered so DMA overlaps WMMA.  Needs M%64==0, N%64==0, K%32==0.
// ---------------------------------------------------------------------------
__global__ __launch_bounds__(128)
void gemm_bf16_wmma(const bf16* __restrict__ A, const bf16* __restrict__ Bt,
                    float* __restrict__ C, int M, int N, int K) {
  __shared__ bf16 sA[2][64][32];
  __shared__ bf16 sB[2][64][32];
  const int tid  = threadIdx.x;
  const int lane = tid & 31;
  const int wave = tid >> 5;
  const int gm0 = blockIdx.y << 6;
  const int gn0 = blockIdx.x << 6;
  const int r  = lane & 15;
  const int ho = (lane & 16) ? 8  : 0;   // A-frag K sub-offset (elements)
  const int hk = (lane & 16) ? 16 : 0;   // unused; B-frags use ho like A
  (void)M;

  const unsigned sA0 = lds_off(&sA[0][0][0]);
  const unsigned sB0 = lds_off(&sB[0][0][0]);

  // Stage one 64x32 A tile + 64x32 B tile into buffer `buf` for K offset k0.
  // 256 16B-chunks per operand; 128 threads x 2 chunks; 4 async instr / wave.
  auto stage = [&](int buf, int k0) {
#pragma unroll
    for (int cc = 0; cc < 2; ++cc) {
      int c    = tid + cc * 128;
      int row  = c >> 2;              // 0..63
      int boff = (c & 3) * 16;        // byte offset within 64B row
      const char* ga = (const char*)(A  + (size_t)(gm0 + row) * K + k0) + boff;
      const char* gb = (const char*)(Bt + (size_t)(gn0 + row) * K + k0) + boff;
      unsigned la = sA0 + (unsigned)(buf * 4096 + row * 64 + boff);
      unsigned lb = sB0 + (unsigned)(buf * 4096 + row * 64 + boff);
      async_copy_b128(la, ga);
      async_copy_b128(lb, gb);
    }
  };

  v8f acc0 = {}, acc1 = {}, acc2 = {}, acc3 = {};
  const int nsteps = K >> 5;

  stage(0, 0);
  for (int i = 0; i < nsteps; ++i) {
    const int buf = i & 1;
    if (i + 1 < nsteps) {
      stage(buf ^ 1, (i + 1) << 5);
      // 4 newer (next-stage) loads may stay in flight; older stage-i done.
      asm volatile("s_wait_asynccnt 0x4" ::: "memory");
    } else {
      asm volatile("s_wait_asynccnt 0x0" ::: "memory");
    }
    __syncthreads();   // all waves' stage-i data visible in LDS

    v16bf a  = load_afrag(&sA[buf][wave * 16 + r][ho]);
    v16bf b0 = load_afrag(&sB[buf][r][ho]);
    v16bf b1 = load_afrag(&sB[buf][16 + r][ho]);
    v16bf b2 = load_afrag(&sB[buf][32 + r][ho]);
    v16bf b3 = load_afrag(&sB[buf][48 + r][ho]);
    acc0 = WMMA_BF16(a, b0, acc0);
    acc1 = WMMA_BF16(a, b1, acc1);
    acc2 = WMMA_BF16(a, b2, acc2);
    acc3 = WMMA_BF16(a, b3, acc3);

    __syncthreads();   // reads done before next iteration overwrites `buf`
  }

  // C/D layout: VGPR v, lanes 0-15 -> row m0+v col lane; lanes 16-31 -> +8.
  const int m0 = gm0 + wave * 16;
  float* crow = C + (size_t)(m0 + ho) * N + gn0 + r;
#pragma unroll
  for (int v = 0; v < 8; ++v) {
    crow[0]  = acc0[v];
    crow[16] = acc1[v];
    crow[32] = acc2[v];
    crow[48] = acc3[v];
    crow += N;
  }
}

// ---------------------------------------------------------------------------
// Per-head RMSNorm + RoPE for q/k, transpose-store for v.
// grid = (T, 24): y in [0,16) q heads, [16,20) k heads, [20,24) v heads.
// qf [T, 2048], kf/vf [T, 512].  Outputs:
//   qb [B,Hq,S,D] bf16, kb [B,Hkv,S,D] bf16, vbT [B,Hkv,D,S] bf16.
// ---------------------------------------------------------------------------
__global__ __launch_bounds__(128)
void qkv_post_kernel(const float* __restrict__ qf, const float* __restrict__ kf,
                     const float* __restrict__ vf, const float* __restrict__ qnw,
                     const float* __restrict__ knw, bf16* __restrict__ qb,
                     bf16* __restrict__ kb, bf16* __restrict__ vbT, int S) {
  const int t = blockIdx.x;
  const int b = t / S, s = t - b * S;
  const int hid = blockIdx.y;
  const int d = threadIdx.x;            // 0..127

  if (hid >= 20) {                      // v head: just transpose + bf16 cast
    int h = hid - 20;
    float v = vf[(size_t)t * 512 + h * 128 + d];
    vbT[((size_t)(b * 4 + h) * 128 + d) * S + s] = (bf16)v;
    return;
  }
  __shared__ float xs[128];
  __shared__ float red[4];
  const bool isq = hid < 16;
  const int h = isq ? hid : hid - 16;
  float x = isq ? qf[(size_t)t * 2048 + h * 128 + d]
                : kf[(size_t)t * 512 + h * 128 + d];
  xs[d] = x;
  float ssq = x * x;
  ssq = wave_sum32(ssq);
  if ((threadIdx.x & 31) == 0) red[threadIdx.x >> 5] = ssq;
  __syncthreads();
  float inv = rsqrtf((red[0] + red[1] + red[2] + red[3]) / 128.f + 1e-6f);
  const float* nw = isq ? qnw : knw;
  float xn = x * inv * nw[d];
  int pd = (d < 64) ? d + 64 : d - 64;
  float xpn = xs[pd] * inv * nw[pd];
  float rot = (d < 64) ? -xpn : xpn;
  int fi = d & 63;
  float invf = __powf(1000000.0f, -((float)(2 * fi)) / 128.0f);
  float ang = (float)s * invf;
  float sn, cs;
  sincosf(ang, &sn, &cs);
  float o = xn * cs + rot * sn;
  if (isq) qb[((size_t)(b * 16 + h) * S + s) * 128 + d] = (bf16)o;
  else     kb[((size_t)(b * 4  + h) * S + s) * 128 + d] = (bf16)o;
}

// ---------------------------------------------------------------------------
// Flash attention with WMMA.  One wave per (b, h, 16-query-row tile).
// Qb [B,Hq,S,D], Kb [B,Hkv,S,D], Vt [B,Hkv,D,S] -> ctx [B*S, Hq*D] bf16.
// Per key tile: 4 WMMA (QK^T) + 8 WMMA (PV), online softmax in f32.
// ---------------------------------------------------------------------------
__global__ __launch_bounds__(32)
void attn_fa_wmma(const bf16* __restrict__ Qb, const bf16* __restrict__ Kb,
                  const bf16* __restrict__ Vt, bf16* __restrict__ ctx, int S) {
  const int qt = blockIdx.x;            // query tile (16 rows)
  const int h  = blockIdx.y;            // 0..15
  const int b  = blockIdx.z;
  const int hk = h >> 2;                // GQA: rep = 4
  const int lane = threadIdx.x;
  const int r  = lane & 15;
  const int ho = (lane & 16) ? 8  : 0;
  const int hkk = (lane & 16) ? 16 : 0;
  const float scale = 0.08838834764831845f;   // 1/sqrt(128)

  const bf16* qrow = Qb + ((size_t)(b * 16 + h) * S + qt * 16 + r) * 128 + ho;
  v16bf qfr[4];
#pragma unroll
  for (int kk = 0; kk < 4; ++kk) qfr[kk] = load_afrag(qrow + kk * 32);

  v8f zero8 = {};
  v8f o[8];
#pragma unroll
  for (int nt = 0; nt < 8; ++nt) o[nt] = zero8;
  float mrun[8], lrun[8];
#pragma unroll
  for (int v = 0; v < 8; ++v) { mrun[v] = -1e30f; lrun[v] = 0.f; }

  __shared__ float pmat[16][17];

  const bf16* kbase = Kb + (size_t)(b * 4 + hk) * S * 128;
  const bf16* vbase = Vt + (size_t)(b * 4 + hk) * 128 * S;

  for (int kt = 0; kt <= qt; ++kt) {
    // ---- S tile = Q (16x128) . K^T (128x16) ----
    v8f sacc = zero8;
#pragma unroll
    for (int kk = 0; kk < 4; ++kk) {
      v16bf kfrag = load_afrag(kbase + (size_t)(kt * 16 + r) * 128 + kk * 32 + ho);
      sacc = WMMA_BF16(qfr[kk], kfrag, sacc);
    }
    const bool diag = (kt == qt);
    float pv[8], sf[8];
#pragma unroll
    for (int v = 0; v < 8; ++v) {
      float sv = sacc[v] * scale;                 // row = v+ho, col = r
      if (diag && r > v + ho) sv = -1e30f;        // causal mask
      float rm = half_max16(sv);
      float mnew = fmaxf(mrun[v], rm);
      sf[v] = __expf(mrun[v] - mnew);
      mrun[v] = mnew;
      float p = __expf(sv - mnew);
      pv[v] = p;
      lrun[v] = lrun[v] * sf[v] + half_sum16(p);
    }
#pragma unroll
    for (int nt = 0; nt < 8; ++nt)
#pragma unroll
      for (int v = 0; v < 8; ++v) o[nt][v] = o[nt][v] * sf[v];

    // ---- stage P (16x16) through LDS into A-fragment layout ----
    __syncthreads();
#pragma unroll
    for (int v = 0; v < 8; ++v) pmat[v + ho][r] = pv[v];
    __syncthreads();
    v16bf pa = {};                                 // lanes 0-15 hold K 0..15; K 16..31 zero-padded
    if (lane < 16) {
#pragma unroll
      for (int j = 0; j < 16; ++j) pa[j] = (bf16)pmat[r][j];
    }

    // ---- O += P (16x16, K-padded to 32) . V (16x128) ----
#pragma unroll
    for (int nt = 0; nt < 8; ++nt) {
      v16bf vb = {};
      if (lane < 16)                               // B-frag lanes 0-15 hold K 0..15
        vb = *(const v16bf*)(vbase + (size_t)(nt * 16 + r) * S + kt * 16);
      o[nt] = WMMA_BF16(pa, vb, o[nt]);
    }
  }

  // ---- epilogue: normalize and store ctx ----
#pragma unroll
  for (int nt = 0; nt < 8; ++nt) {
#pragma unroll
    for (int v = 0; v < 8; ++v) {
      float val = o[nt][v] / lrun[v];
      size_t row = (size_t)b * S + qt * 16 + v + ho;
      ctx[row * 2048 + h * 128 + nt * 16 + r] = (bf16)val;
    }
  }
}

// ---------------------------------------------------------------------------
// Router: logits = xn @ router_w [H,8]; softmax; top-2; normalized dense wts.
// ---------------------------------------------------------------------------
__global__ __launch_bounds__(256)
void router_topk_kernel(const bf16* __restrict__ xn, const float* __restrict__ rw,
                        float* __restrict__ wout, int H) {
  const int t = blockIdx.x;
  const int lane = threadIdx.x & 31;
  const int wv = threadIdx.x >> 5;      // expert id
  float p = 0.f;
  for (int i = lane; i < H; i += 32)
    p += (float)xn[(size_t)t * H + i] * rw[(size_t)i * 8 + wv];
  p = wave_sum32(p);
  __shared__ float lg[8];
  if (lane == 0) lg[wv] = p;
  __syncthreads();
  if (threadIdx.x == 0) {
    float mx = -1e30f;
#pragma unroll
    for (int e = 0; e < 8; ++e) mx = fmaxf(mx, lg[e]);
    float pe[8]; float sum = 0.f;
#pragma unroll
    for (int e = 0; e < 8; ++e) { pe[e] = __expf(lg[e] - mx); sum += pe[e]; }
#pragma unroll
    for (int e = 0; e < 8; ++e) pe[e] /= sum;
    int i1 = 0;
#pragma unroll
    for (int e = 1; e < 8; ++e) if (pe[e] > pe[i1]) i1 = e;
    int i2 = (i1 == 0) ? 1 : 0;
#pragma unroll
    for (int e = 0; e < 8; ++e) if (e != i1 && pe[e] > pe[i2]) i2 = e;
    float s2 = pe[i1] + pe[i2];
#pragma unroll
    for (int e = 0; e < 8; ++e) wout[(size_t)t * 8 + e] = 0.f;
    wout[(size_t)t * 8 + i1] = pe[i1] / s2;
    wout[(size_t)t * 8 + i2] = pe[i2] / s2;
  }
}

// act = silu(g) * u   with gu = [T, 2I], g first I cols, u last I cols.
__global__ __launch_bounds__(256)
void silu_mul_kernel(const float* __restrict__ gu, bf16* __restrict__ act, int I) {
  size_t idx = (size_t)blockIdx.x * 256 + threadIdx.x;
  size_t t = idx / I;
  size_t i = idx - t * I;
  float g = gu[t * 2 * I + i];
  float u = gu[t * 2 * I + I + i];
  float sg = g / (1.f + __expf(-g));
  act[idx] = (bf16)(sg * u);
}

// moe += w[t,e] * y   (e==0 initializes)
__global__ __launch_bounds__(256)
void moe_accum_kernel(const float* __restrict__ y, const float* __restrict__ wts,
                      float* __restrict__ moe, int e, int H) {
  size_t idx = (size_t)blockIdx.x * 256 + threadIdx.x;
  size_t t = idx / H;
  float val = wts[t * 8 + e] * y[idx];
  moe[idx] = (e == 0) ? val : (moe[idx] + val);
}

__global__ __launch_bounds__(256)
void final_add_kernel(const float* __restrict__ resid, const float* __restrict__ moe,
                      float* __restrict__ out) {
  size_t idx = (size_t)blockIdx.x * 256 + threadIdx.x;
  out[idx] = resid[idx] + moe[idx];
}

// ---------------------------------------------------------------------------
// Host launcher
// ---------------------------------------------------------------------------
static inline void gemm_launch(const bf16* A, const bf16* Bt, float* C,
                               int M, int N, int K, hipStream_t s) {
  dim3 grid(N / 64, M / 64);
  gemm_bf16_wmma<<<grid, 128, 0, s>>>(A, Bt, C, M, N, K);
}

extern "C" void kernel_launch(void* const* d_in, const int* in_sizes, int n_in,
                              void* d_out, int out_size, void* d_ws, size_t ws_size,
                              hipStream_t stream) {
  (void)in_sizes; (void)n_in; (void)out_size; (void)ws_size;

  const float* hidden    = (const float*)d_in[0];
  const float* ln1_w     = (const float*)d_in[1];
  const float* ln2_w     = (const float*)d_in[2];
  const float* q_w       = (const float*)d_in[3];
  const float* k_w       = (const float*)d_in[4];
  const float* v_w       = (const float*)d_in[5];
  const float* o_w       = (const float*)d_in[6];
  const float* q_norm_w  = (const float*)d_in[7];
  const float* k_norm_w  = (const float*)d_in[8];
  const float* router_w  = (const float*)d_in[9];
  const float* gate_up_w = (const float*)d_in[10];
  const float* down_w    = (const float*)d_in[11];

  const int B = 2, S = 1024, H = 1024, T = B * S;
  const int D = 128, HQ = 16, HKV = 4, I = 1024;
  const int QN = HQ * D;    // 2048
  const int KN = HKV * D;   // 512
  (void)D;

  char* ws = (char*)d_ws;
  size_t off = 0;
  auto carve = [&](size_t bytes) -> void* {
    void* p = ws + off;
    off = (off + bytes + 255) & ~(size_t)255;
    return p;
  };

  // bf16 transposed weights [N,K]
  bf16* qw_t  = (bf16*)carve((size_t)QN * H * 2);
  bf16* kw_t  = (bf16*)carve((size_t)KN * H * 2);
  bf16* vw_t  = (bf16*)carve((size_t)KN * H * 2);
  bf16* ow_t  = (bf16*)carve((size_t)H * QN * 2);
  bf16* guw_t = (bf16*)carve((size_t)8 * 2 * I * H * 2);
  bf16* dw_t  = (bf16*)carve((size_t)8 * H * I * 2);
  // activations
  bf16*  xn1  = (bf16*)carve((size_t)T * H * 2);
  float* qf   = (float*)carve((size_t)T * QN * 4);   // reused as gu in MoE loop
  float* kf   = (float*)carve((size_t)T * KN * 4);
  float* vf   = (float*)carve((size_t)T * KN * 4);
  bf16*  qb   = (bf16*)carve((size_t)T * QN * 2);
  bf16*  kb   = (bf16*)carve((size_t)T * KN * 2);
  bf16*  vbt  = (bf16*)carve((size_t)T * KN * 2);
  bf16*  ctxb = (bf16*)carve((size_t)T * QN * 2);
  float* attn_out = (float*)carve((size_t)T * H * 4);
  float* resid    = (float*)carve((size_t)T * H * 4);
  bf16*  xn2      = (bf16*)carve((size_t)T * H * 2);
  float* rwts     = (float*)carve((size_t)T * 8 * 4);
  bf16*  act      = (bf16*)carve((size_t)T * I * 2);
  float* yb       = (float*)carve((size_t)T * H * 4);
  float* moe      = (float*)carve((size_t)T * H * 4);
  float* gu       = qf;   // alias: qf dead after qkv_post

  // ---- weight conversion (transpose to [N,K] bf16) ----
  transpose_to_bf16<<<dim3((H * QN) / 256, 1, 1), 256, 0, stream>>>(q_w, qw_t, H, QN);
  transpose_to_bf16<<<dim3((H * KN) / 256, 1, 1), 256, 0, stream>>>(k_w, kw_t, H, KN);
  transpose_to_bf16<<<dim3((H * KN) / 256, 1, 1), 256, 0, stream>>>(v_w, vw_t, H, KN);
  transpose_to_bf16<<<dim3((QN * H) / 256, 1, 1), 256, 0, stream>>>(o_w, ow_t, QN, H);
  transpose_to_bf16<<<dim3((H * 2 * I) / 256, 1, 8), 256, 0, stream>>>(gate_up_w, guw_t, H, 2 * I);
  transpose_to_bf16<<<dim3((I * H) / 256, 1, 8), 256, 0, stream>>>(down_w, dw_t, I, H);

  // ---- attention ----
  rmsnorm_bf16_kernel<<<T, 256, 0, stream>>>(hidden, ln1_w, xn1, H);
  gemm_launch(xn1, qw_t, qf, T, QN, H, stream);
  gemm_launch(xn1, kw_t, kf, T, KN, H, stream);
  gemm_launch(xn1, vw_t, vf, T, KN, H, stream);
  qkv_post_kernel<<<dim3(T, 24), 128, 0, stream>>>(qf, kf, vf, q_norm_w, k_norm_w,
                                                   qb, kb, vbt, S);
  attn_fa_wmma<<<dim3(S / 16, HQ, B), 32, 0, stream>>>(qb, kb, vbt, ctxb, S);
  gemm_launch(ctxb, ow_t, attn_out, T, H, QN, stream);

  // ---- post-attn norm ----
  add_resid_rmsnorm_kernel<<<T, 256, 0, stream>>>(hidden, attn_out, ln2_w, resid, xn2, H);

  // ---- MoE ----
  router_topk_kernel<<<T, 256, 0, stream>>>(xn2, router_w, rwts, H);
  for (int e = 0; e < 8; ++e) {
    gemm_launch(xn2, guw_t + (size_t)e * 2 * I * H, gu, T, 2 * I, H, stream);
    silu_mul_kernel<<<((size_t)T * I) / 256, 256, 0, stream>>>(gu, act, I);
    gemm_launch(act, dw_t + (size_t)e * H * I, yb, T, H, I, stream);
    moe_accum_kernel<<<((size_t)T * H) / 256, 256, 0, stream>>>(yb, rwts, moe, e, H);
  }
  final_add_kernel<<<((size_t)T * H) / 256, 256, 0, stream>>>(resid, moe, (float*)d_out);
}
